// GraphDocEmbeddings_15367392985620
// MI455X (gfx1250) — compile-verified, matrix operand order checked by reading, not run on hardware
//
#include <hip/hip_runtime.h>
#include <math.h>

#define B_    2
#define L_    1024
#define K_    50
#define HID_  768
#define PE_   384
#define COORD_ 128
#define TBL_  1024   // sinusoidal INIT size (mod wrap)

typedef __attribute__((ext_vector_type(16))) _Float16 v16h;
typedef __attribute__((ext_vector_type(8)))  _Float16 v8h;
typedef __attribute__((ext_vector_type(8)))  float    v8f;

// ------------------------------------------------------------------
// 1) sinusoidal table in f16:  sh[n][j], n<1024, j<384
//    j<192: sin(n*freq[j]); j>=192: cos(n*freq[j-192]); row 0 zeroed
// ------------------------------------------------------------------
__global__ __launch_bounds__(256) void k_sin(_Float16* __restrict__ sh) {
  int idx = blockIdx.x * 256 + threadIdx.x;        // 1024*384 = 393216
  if (idx >= TBL_ * PE_) return;
  int n = idx / PE_, j = idx % PE_;
  float v = 0.f;
  if (n != 0) {
    int jj = (j >= 192) ? (j - 192) : j;
    float freq = __expf(-(float)jj * (logf(10000.0f) / 191.0f));
    float ang  = (float)n * freq;
    v = (j >= 192) ? cosf(ang) : sinf(ang);
  }
  sh[idx] = (_Float16)v;
}

// ------------------------------------------------------------------
// 2) combine 4 weight matrices into 4 half-width f16 matrices + bias sum
//    Wh[t][o][c], t: 0=x1(tl+bl lo), 1=x2(tr+br lo), 2=y1(tl+tr hi), 3=y2(bl+br hi)
// ------------------------------------------------------------------
__global__ __launch_bounds__(256) void k_comb(
    const float* __restrict__ Wtl, const float* __restrict__ Wtr,
    const float* __restrict__ Wbl, const float* __restrict__ Wbr,
    const float* __restrict__ btl, const float* __restrict__ btr,
    const float* __restrict__ bbl, const float* __restrict__ bbr,
    _Float16* __restrict__ Wh, float* __restrict__ bias_sum) {
  int idx = blockIdx.x * 256 + threadIdx.x;        // 4*768*384 = 1179648
  if (idx < HID_) bias_sum[idx] = btl[idx] + btr[idx] + bbl[idx] + bbr[idx];
  if (idx >= 4 * HID_ * PE_) return;
  int t = idx / (HID_ * PE_);
  int r = idx % (HID_ * PE_);
  int o = r / PE_, c = r % PE_;
  float v;
  if      (t == 0) v = Wtl[o * 768 + c]       + Wbl[o * 768 + c];
  else if (t == 1) v = Wtr[o * 768 + c]       + Wbr[o * 768 + c];
  else if (t == 2) v = Wtl[o * 768 + 384 + c] + Wtr[o * 768 + 384 + c];
  else             v = Wbl[o * 768 + 384 + c] + Wbr[o * 768 + 384 + c];
  Wh[idx] = (_Float16)v;
}

// ------------------------------------------------------------------
// 3) WMMA GEMM:  P[t][d][o] = sum_c sin[d][c] * Wh[t][o][c]
//    M=1024, N=768, K=384. One wave per 16x16 tile, f16 inputs, f32 acc.
// ------------------------------------------------------------------
__global__ __launch_bounds__(32) void k_gemm(
    const _Float16* __restrict__ A,   // [1024][384]
    const _Float16* __restrict__ W,   // [4][768][384]
    float* __restrict__ P) {          // [4][1024][768]
  const int tilesM = TBL_ / 16, tilesN = HID_ / 16;   // 64, 48
  int tile = blockIdx.x;                              // 4*64*48 = 12288
  int t    = tile / (tilesM * tilesN);
  int rem  = tile % (tilesM * tilesN);
  int mt   = rem / tilesN;
  int nt   = rem % tilesN;
  int lane = threadIdx.x;
  int hi   = (lane >= 16) ? 1 : 0;
  int l15  = lane & 15;
  int koff = hi ? 8 : 0;

  const _Float16* arow = A + (size_t)(mt * 16 + l15) * PE_ + koff;
  const _Float16* brow = W + (size_t)t * HID_ * PE_ + (size_t)(nt * 16 + l15) * PE_ + koff;

  v8f acc = {};
#pragma unroll
  for (int kb = 0; kb < PE_; kb += 32) {
    v8h a0 = *(const v8h*)(arow + kb);
    v8h a1 = *(const v8h*)(arow + kb + 16);
    v8h b0 = *(const v8h*)(brow + kb);
    v8h b1 = *(const v8h*)(brow + kb + 16);
    v16h av, bv;
#pragma unroll
    for (int i = 0; i < 8; ++i) { av[i] = a0[i]; av[8 + i] = a1[i];
                                  bv[i] = b0[i]; bv[8 + i] = b1[i]; }
    acc = __builtin_amdgcn_wmma_f32_16x16x32_f16(false, av, false, bv,
                                                 (short)0, acc, false, false);
  }
  float* cbase = P + (size_t)t * TBL_ * HID_;
  int crow = mt * 16 + (hi ? 8 : 0);
  int ccol = nt * 16 + l15;
#pragma unroll
  for (int r = 0; r < 8; ++r) cbase[(size_t)(crow + r) * HID_ + ccol] = acc[r];
}

// ------------------------------------------------------------------
// 4) box centers (with attention-mask -> BIG fill)
// ------------------------------------------------------------------
__global__ __launch_bounds__(256) void k_centers(const int* __restrict__ bbox,
                                                 const int* __restrict__ amask,
                                                 float2* __restrict__ ctr) {
  int idx = blockIdx.x * 256 + threadIdx.x;        // B*L = 2048
  if (idx >= B_ * L_) return;
  const int4 bb = ((const int4*)bbox)[idx];
  float x0, y0, x1, y1;
  if (amask[idx] == 0) { x0 = y0 = x1 = y1 = 100000000.0f; }
  else { x0 = (float)bb.x; y0 = (float)bb.y; x1 = (float)bb.z; y1 = (float)bb.w; }
  ctr[idx] = make_float2((x0 + x1) * 0.5f, (y0 + y1) * 0.5f);
}

// ------------------------------------------------------------------
// 5) KNN: one wave per (b,i). 32 dists/lane in regs, 50x u64 min-reduce.
//    Writes topk_index, local_mask, and packed diff indices for gather.
// ------------------------------------------------------------------
__global__ __launch_bounds__(32) void k_knn(const int* __restrict__ bbox,
                                            const float2* __restrict__ ctr,
                                            int* __restrict__ topk_out,
                                            int* __restrict__ mask_out,
                                            int4* __restrict__ pidx) {
  int row  = blockIdx.x;                 // 0..B*L-1
  int b    = row / L_;
  int lane = threadIdx.x;
  const float2 ci = ctr[row];
  const float2* cb = ctr + (size_t)b * L_;

  float dist[32];
#pragma unroll
  for (int r = 0; r < 32; ++r) {
    int j = r * 32 + lane;
    float2 cj = cb[j];
    float dx = ci.x - cj.x, dy = ci.y - cj.y;
    dist[r] = dx * dx + dy * dy;
  }

  int* mrow = mask_out + (size_t)row * L_;
#pragma unroll
  for (int r = 0; r < 32; ++r) mrow[r * 32 + lane] = 0;

  const int4 bi = ((const int4*)bbox)[row];
  const int4* bb = (const int4*)bbox + (size_t)b * L_;
  const float INF = __builtin_inff();

  for (int k = 0; k < K_; ++k) {
    float best = INF;
    unsigned bidx = 0xFFFFFFFFu;
#pragma unroll
    for (int r = 0; r < 32; ++r)
      if (dist[r] < best) { best = dist[r]; bidx = (unsigned)(r * 32 + lane); }
    unsigned long long key =
        ((unsigned long long)__float_as_uint(best) << 32) | (unsigned long long)bidx;
#pragma unroll
    for (int off = 16; off > 0; off >>= 1) {
      unsigned long long o = __shfl_xor(key, off, 32);
      key = (o < key) ? o : key;
    }
    unsigned jstar = (unsigned)(key & 0xFFFFFFFFu);
    int slot = (int)(jstar >> 5), ln = (int)(jstar & 31);
    if (lane == ln) {
#pragma unroll
      for (int r = 0; r < 32; ++r) if (r == slot) dist[r] = INF;
    }
    if (lane == 0) {
      topk_out[(size_t)row * K_ + k] = (int)jstar;
      mrow[jstar] = 1;
      int4 bj = bb[jstar];
      int d0 = min(max(bi.x - bj.x, -1000), 1000) & 1023;  // x1 diff
      int d1 = min(max(bi.z - bj.z, -1000), 1000) & 1023;  // x2 diff
      int d2 = min(max(bi.y - bj.y, -1000), 1000) & 1023;  // y1 diff
      int d3 = min(max(bi.w - bj.w, -1000), 1000) & 1023;  // y2 diff
      pidx[(size_t)row * K_ + k] = make_int4(d0, d1, d2, d3);
    }
  }
}

// ------------------------------------------------------------------
// 6) layout embedding gather + LayerNorm. One block(256) per (b,i).
// ------------------------------------------------------------------
__global__ __launch_bounds__(256) void k_layout(const int* __restrict__ bbox,
                                                const float* __restrict__ xt,
                                                const float* __restrict__ yt,
                                                const float* __restrict__ ht,
                                                const float* __restrict__ wt,
                                                const float* __restrict__ lng,
                                                const float* __restrict__ lnb,
                                                float* __restrict__ out) {
  int row = blockIdx.x;
  int tid = threadIdx.x;
  const int4 bb = ((const int4*)bbox)[row];
  float v[3]; float s = 0.f, s2 = 0.f;
#pragma unroll
  for (int q = 0; q < 3; ++q) {
    int o = tid + q * 256;
    int seg = o >> 7, c = o & 127;
    float x;
    if      (seg == 0) x = xt[(size_t)bb.x * COORD_ + c];
    else if (seg == 1) x = yt[(size_t)bb.y * COORD_ + c];
    else if (seg == 2) x = xt[(size_t)bb.z * COORD_ + c];
    else if (seg == 3) x = yt[(size_t)bb.w * COORD_ + c];
    else if (seg == 4) x = ht[(size_t)(bb.w - bb.y) * COORD_ + c];
    else               x = wt[(size_t)(bb.z - bb.x) * COORD_ + c];
    v[q] = x; s += x; s2 += x * x;
  }
  __shared__ float ss[256], sq[256];
  ss[tid] = s; sq[tid] = s2; __syncthreads();
  for (int st = 128; st > 0; st >>= 1) {
    if (tid < st) { ss[tid] += ss[tid + st]; sq[tid] += sq[tid + st]; }
    __syncthreads();
  }
  float mean = ss[0] * (1.0f / HID_);
  float var  = sq[0] * (1.0f / HID_) - mean * mean;
  float rstd = rsqrtf(var + 1e-12f);
  float* orow = out + (size_t)row * HID_;
#pragma unroll
  for (int q = 0; q < 3; ++q) {
    int o = tid + q * 256;
    orow[o] = (v[q] - mean) * rstd * lng[o] + lnb[o];
  }
}

// ------------------------------------------------------------------
// 7) p = P0[dx1] + P1[dx2] + P2[dy1] + P3[dy2] + bias  (float4 gather-add)
// ------------------------------------------------------------------
__global__ __launch_bounds__(256) void k_pgather(const float4* __restrict__ P4,  // [4][1024][192]
                                                 const int4*   __restrict__ pidx,
                                                 const float4* __restrict__ bias4,
                                                 float4* __restrict__ out4) {
  const int R4 = HID_ / 4;                              // 192
  size_t idx = (size_t)blockIdx.x * 256 + threadIdx.x;  // B*L*K*192
  size_t rowi = idx / R4;
  int    o4   = (int)(idx % R4);
  int4 d = pidx[rowi];
  const size_t T = (size_t)TBL_ * R4;                   // per-table float4 count
  float4 a = P4[0 * T + (size_t)d.x * R4 + o4];
  float4 b = P4[1 * T + (size_t)d.y * R4 + o4];
  float4 c = P4[2 * T + (size_t)d.z * R4 + o4];
  float4 e = P4[3 * T + (size_t)d.w * R4 + o4];
  float4 g = bias4[o4];
  float4 r;
  r.x = a.x + b.x + c.x + e.x + g.x;
  r.y = a.y + b.y + c.y + e.y + g.y;
  r.z = a.z + b.z + c.z + e.z + g.z;
  r.w = a.w + b.w + c.w + e.w + g.w;
  out4[idx] = r;
}

// ------------------------------------------------------------------
extern "C" void kernel_launch(void* const* d_in, const int* in_sizes, int n_in,
                              void* d_out, int out_size, void* d_ws, size_t ws_size,
                              hipStream_t stream) {
  const int*   bbox  = (const int*)d_in[0];
  const int*   amask = (const int*)d_in[1];
  const float* x_tab = (const float*)d_in[2];
  const float* y_tab = (const float*)d_in[3];
  const float* h_tab = (const float*)d_in[4];
  const float* w_tab = (const float*)d_in[5];
  const float* Wtl   = (const float*)d_in[6];
  const float* btl   = (const float*)d_in[7];
  const float* Wtr   = (const float*)d_in[8];
  const float* btr   = (const float*)d_in[9];
  const float* Wbl   = (const float*)d_in[10];
  const float* bbl   = (const float*)d_in[11];
  const float* Wbr   = (const float*)d_in[12];
  const float* bbr   = (const float*)d_in[13];
  const float* lng   = (const float*)d_in[14];
  const float* lnb   = (const float*)d_in[15];

  // ---- output layout (elements) ----
  float* out_layout = (float*)d_out;                                   // 2*1024*768
  float* out_p      = out_layout + (size_t)B_ * L_ * HID_;             // 2*1024*50*768
  int*   out_topk   = (int*)(out_p + (size_t)B_ * L_ * K_ * HID_);     // 2*1024*50
  int*   out_mask   = out_topk + (size_t)B_ * L_ * K_;                 // 2*1024*1024

  // ---- workspace layout (bytes) ----
  char* ws = (char*)d_ws;
  float2*   ctr      = (float2*)ws;                       size_t off = (size_t)B_ * L_ * sizeof(float2);
  _Float16* sin_h    = (_Float16*)(ws + off);             off += (size_t)TBL_ * PE_ * sizeof(_Float16);
  _Float16* Wh       = (_Float16*)(ws + off);             off += (size_t)4 * HID_ * PE_ * sizeof(_Float16);
  float*    bias_sum = (float*)(ws + off);                off += (size_t)HID_ * sizeof(float);
  float*    P        = (float*)(ws + off);                off += (size_t)4 * TBL_ * HID_ * sizeof(float);
  int4*     pidx     = (int4*)(ws + off);                 off += (size_t)B_ * L_ * K_ * sizeof(int4);
  (void)ws_size; (void)n_in; (void)in_sizes; (void)out_size;

  // 1) sinusoidal table (f16)
  k_sin<<<(TBL_ * PE_ + 255) / 256, 256, 0, stream>>>(sin_h);

  // 2) combined weights (f16) + bias sum
  k_comb<<<(4 * HID_ * PE_ + 255) / 256, 256, 0, stream>>>(
      Wtl, Wtr, Wbl, Wbr, btl, btr, bbl, bbr, Wh, bias_sum);

  // 3) WMMA precompute: P[t] = sin_tab @ Wh[t]^T
  k_gemm<<<4 * (TBL_ / 16) * (HID_ / 16), 32, 0, stream>>>(sin_h, Wh, P);

  // 4) centers
  k_centers<<<(B_ * L_ + 255) / 256, 256, 0, stream>>>(bbox, amask, ctr);

  // 5) knn + mask + gather indices
  k_knn<<<B_ * L_, 32, 0, stream>>>(bbox, ctr, out_topk, out_mask, pidx);

  // 6) layout embedding + LayerNorm
  k_layout<<<B_ * L_, 256, 0, stream>>>(bbox, x_tab, y_tab, h_tab, w_tab, lng, lnb, out_layout);

  // 7) p gather-add
  size_t total4 = (size_t)B_ * L_ * K_ * (HID_ / 4);     // 19,660,800
  k_pgather<<<(unsigned)(total4 / 256), 256, 0, stream>>>(
      (const float4*)P, pidx, (const float4*)bias_sum, (float4*)out_p);
}
